// Net_31834297598780
// MI455X (gfx1250) — matrix operand order repaired and checked by reading, not compiled, not run on hardware
//
#include <hip/hip_runtime.h>
#include <hip/hip_bf16.h>

#define NDIM 87
#define NP   96      // padded N/K
#define SP   98      // LDS row stride (bf16 elems) for 96-wide arrays (49 dwords)
#define WCOLS 174
#define WS   194     // LDS row stride for W (97 dwords)
#define TILES 6      // 96/16

typedef __bf16 v16bf __attribute__((ext_vector_type(16)));
typedef float  v8f   __attribute__((ext_vector_type(8)));

union Frag { v16bf h; unsigned int u[8]; };

__device__ __forceinline__ unsigned short f2bf(float f) {
    // round-to-nearest-even f32 -> bf16 (inputs are finite)
    unsigned int u = __float_as_uint(f);
    u += 0x7FFFu + ((u >> 16) & 1u);
    return (unsigned short)(u >> 16);
}

__global__ __launch_bounds__(256) void diffusion_net_kernel(
    const float* __restrict__ x,
    const float* __restrict__ evals,
    const float* __restrict__ evecs,
    const float* __restrict__ dtim,
    const float* __restrict__ W,
    const float* __restrict__ bias,
    float* __restrict__ out)
{
    __shared__ unsigned short sEv[NP * SP];   // evecs (bf16, B-operand of stage 1)
    __shared__ unsigned short sA [NP * SP];   // evecs * exp(-t_i * eval_m)
    __shared__ unsigned short sX [NP * SP];   // x (bf16)
    __shared__ unsigned short sK [NP * SP];   // stage-1 result (bf16)
    __shared__ unsigned short sW [NP * WS];   // W (bf16, row-major = B-operand of stage 2)
    __shared__ float sEval[NP];
    __shared__ float sT[NP];
    __shared__ float sBias[NP];

    const int b   = blockIdx.x;
    const int tid = threadIdx.x;
    const size_t baseBN = (size_t)b * NDIM * NDIM;

    // ---- small per-batch vectors ----
    for (int i = tid; i < NP; i += 256) {
        sEval[i] = (i < NDIM) ? evals[(size_t)b * NDIM + i] : 0.0f;
        sT[i]    = (i < NDIM) ? fmaxf(dtim[i], 1e-8f) : 0.0f;
        sBias[i] = (i < NDIM) ? bias[i] : 0.0f;
    }
    __syncthreads();

    // ---- stage operands into LDS as bf16, zero-padded to 96 ----
    for (int idx = tid; idx < NP * NP; idx += 256) {
        const int i = idx / NP, m = idx % NP;
        float e = 0.f, a = 0.f, xv = 0.f;
        if (i < NDIM && m < NDIM) {
            e  = evecs[baseBN + (size_t)i * NDIM + m];
            a  = e * __expf(-sT[i] * sEval[m]);
            xv = x[baseBN + (size_t)i * NDIM + m];
        }
        sEv[i * SP + m] = f2bf(e);
        sA [i * SP + m] = f2bf(a);
        sX [i * SP + m] = f2bf(xv);
    }
    // W is shared across all blocks -> served from the 192MB L2
    for (int idx = tid; idx < NP * WS; idx += 256) {
        const int j = idx / WS, k = idx % WS;
        const float w = (j < NDIM && k < WCOLS) ? W[(size_t)j * WCOLS + k] : 0.0f;
        sW[idx] = f2bf(w);
    }
    __syncthreads();

    const int wave = tid >> 5;
    const int lane = tid & 31;
    const int l16  = lane & 15;
    const int lh   = lane >> 4;

    // ---- stage 1: kernel = (evecs*expa) @ evecs^T, bf16 result into sK ----
    for (int t = wave; t < TILES * TILES; t += 8) {
        const int ti = t / TILES, tj = t % TILES;
        v8f c = {};
        #pragma unroll
        for (int kk = 0; kk < 3; ++kk) {
            Frag af, bfr;
            #pragma unroll
            for (int v = 0; v < 8; ++v) {
                const int ka = 2 * (v & 3) + 8 * lh + 16 * (v >> 2) + 32 * kk;
                af.u[v]  = *(const unsigned int*)&sA [(ti * 16 + l16) * SP + ka];
                const int kb = 2 * (v & 3) + 8 * lh + 16 * (v >> 2) + 32 * kk;
                bfr.u[v] = *(const unsigned int*)&sEv[(tj * 16 + l16) * SP + kb];
            }
            c = __builtin_amdgcn_wmma_f32_16x16x32_bf16(false, af.h, false, bfr.h,
                                                        (short)0, c, false, false);
        }
        #pragma unroll
        for (int r = 0; r < 8; ++r) {
            const int row = ti * 16 + r + 8 * lh;     // D layout: M = r + 8*(lane>=16)
            sK[row * SP + tj * 16 + l16] = f2bf(c[r]);
        }
    }
    __syncthreads();

    // ---- stage 2: out = [x | kernel] @ W^T + bias ----
    for (int t = wave; t < TILES * TILES; t += 8) {
        const int ti = t / TILES, tj = t % TILES;
        v8f c = {};
        #pragma unroll
        for (int kk = 0; kk < 6; ++kk) {
            const unsigned short* aSrc = (kk < 3) ? sX : sK;
            const int ck = (kk < 3) ? kk : (kk - 3);
            Frag af, bfr;
            #pragma unroll
            for (int v = 0; v < 8; ++v) {
                const int ka = 2 * (v & 3) + 8 * lh + 16 * (v >> 2) + 32 * ck;
                af.u[v]  = *(const unsigned int*)&aSrc[(ti * 16 + l16) * SP + ka];
                const int kb = 2 * (v & 3) + 8 * lh + 16 * (v >> 2) + 32 * kk;   // W cols 0..191
                bfr.u[v] = *(const unsigned int*)&sW[(tj * 16 + l16) * WS + kb];
            }
            c = __builtin_amdgcn_wmma_f32_16x16x32_bf16(false, af.h, false, bfr.h,
                                                        (short)0, c, false, false);
        }
        #pragma unroll
        for (int r = 0; r < 8; ++r) {
            const int row = ti * 16 + r + 8 * lh;
            const int col = tj * 16 + l16;
            if (row < NDIM && col < NDIM)
                out[baseBN + (size_t)row * NDIM + col] = c[r] + sBias[col];
        }
    }
}

extern "C" void kernel_launch(void* const* d_in, const int* in_sizes, int n_in,
                              void* d_out, int out_size, void* d_ws, size_t ws_size,
                              hipStream_t stream) {
    const float* x     = (const float*)d_in[0];
    const float* evals = (const float*)d_in[1];
    const float* evecs = (const float*)d_in[2];
    const float* dtim  = (const float*)d_in[3];
    const float* W     = (const float*)d_in[4];
    const float* bias  = (const float*)d_in[5];
    float* out = (float*)d_out;

    const int B = in_sizes[0] / (NDIM * NDIM);   // 4096
    diffusion_net_kernel<<<dim3(B), dim3(256), 0, stream>>>(
        x, evals, evecs, dtim, W, bias, out);
}
